// Memory_88648124991303
// MI455X (gfx1250) — matrix operand order repaired and checked by reading, not compiled, not run on hardware
//
#include <hip/hip_runtime.h>
#include <hip/hip_bf16.h>

typedef __attribute__((ext_vector_type(16))) _Float16 v16h;
typedef __attribute__((ext_vector_type(8)))  float    v8f;

#define NPIX   65536      // 16 * 64 * 64
#define CCH    512        // channels
#define NKEY   150
#define MPAD   160        // 10 tiles of 16
#define NTILE  10
#define NKS    16         // K-steps of 32
#define QB     2097152    // 512*4096 : batch stride in elements
#define CSTR   4096       // channel stride in elements (64*64)
#define FRAG   32         // f16 per (col, kstep) fragment row

// ---------------------------------------------------------------------------
// Kernel 1: f16 WMMA GEMM (raw_q @ keys^T) fused with row-norm + argmin.
// Block = 256 threads (8 waves). Each wave: 16 rows x 160 cols.
// Grid = 512 blocks (512 * 8 waves * 16 rows = 65536 rows).
// LDS (164 KB) limits us to 1 block/WGP = 2 waves/SIMD, so we tell the
// compiler to budget registers for exactly 2 waves/EU: ~80 accum VGPRs +
// 80 B-fragment VGPRs + pipelined A loads fit without spilling, letting
// the fully-unrolled K loop software-pipeline LDS loads under the WMMAs.
// ---------------------------------------------------------------------------
__global__ void
__attribute__((amdgpu_flat_work_group_size(256, 256), amdgpu_waves_per_eu(2)))
gemm_argmin_kernel(const float* __restrict__ q,
                   const float* __restrict__ keys,
                   int* __restrict__ idx_out,
                   float* __restrict__ rinv_out) {
    extern __shared__ char smem[];
    _Float16* lds_keys = (_Float16*)smem;                  // [NKS][MPAD][32]
    float*    lds_k2   = (float*)(smem + NKS * MPAD * FRAG * 2);

    // ---- stage keys -> LDS (f16, fragment-major), zero-pad rows 150..159 ----
    for (int i = threadIdx.x; i < MPAD * CCH; i += blockDim.x) {
        int m = i >> 9, c = i & (CCH - 1);
        float v = (m < NKEY) ? keys[m * CCH + c] : 0.0f;
        int dst = ((c >> 5) * MPAD + m) * FRAG + (c & 31);
        lds_keys[dst] = (_Float16)v;
    }
    // ---- k2[m] = sum keys[m]^2 (f32, from global; keys are L2-resident) ----
    if (threadIdx.x < MPAD) {
        float s = 0.0f;
        if (threadIdx.x < NKEY) {
            const float* kr = keys + threadIdx.x * CCH;
#pragma unroll 8
            for (int c = 0; c < CCH; ++c) s += kr[c] * kr[c];
        }
        lds_k2[threadIdx.x] = s;
    }
    __syncthreads();

    const int lane = threadIdx.x & 31;
    const int wv   = threadIdx.x >> 5;
    const int n0   = blockIdx.x * 128 + wv * 16;   // first row of this wave
    const int b    = n0 >> 12;                     // batch index
    const int p0   = n0 & 4095;                    // pixel offset within batch

    // Per-lane base: row (lane & 15) of this 16-row block.
    const float* qbase = q + (size_t)b * QB + (size_t)(p0 + (lane & 15));
    const int half = lane >> 4;            // 0: lanes 0-15, 1: lanes 16-31

    // Per-lane LDS base for B fragments: + t*1024B per tile, +10240B per kstep
    const _Float16* bbase =
        lds_keys + (((lane & 15) << 5) | (half << 4));

    v8f acc[NTILE];
#pragma unroll
    for (int t = 0; t < NTILE; ++t)
        acc[t] = (v8f){0.f, 0.f, 0.f, 0.f, 0.f, 0.f, 0.f, 0.f};

    float sumsq = 0.0f;   // partial sum of squares of this row's channels

    // ---- software-pipelined A loads: prefetch K-step 0 ----
    // A fragment (contiguous K order, matching B):
    //   lanes 0-15 : K = k0+0..7 (elems 0..7), k0+8..15 (elems 8..15)
    //   lanes 16-31: K = k0+16..23,            k0+24..31
    float x0[8], x1[8];
    {
        const int cb = half << 4;
#pragma unroll
        for (int j = 0; j < 8; ++j) {
            x0[j] = qbase[(size_t)(cb + j)      * CSTR];
            x1[j] = qbase[(size_t)(cb + 8 + j)  * CSTR];
        }
    }

#pragma unroll
    for (int ks = 0; ks < NKS; ++ks) {
        // ---- issue all 10 B-fragment loads (distinct registers) ----
        const _Float16* bk = bbase + ks * (MPAD * FRAG);
        v16h bf[NTILE];
#pragma unroll
        for (int t = 0; t < NTILE; ++t)
            bf[t] = *(const v16h*)(bk + t * (16 * FRAG));

        // ---- pack current A fragment, accumulate row sum-of-squares ----
        v16h a;
#pragma unroll
        for (int j = 0; j < 8; ++j) {
            sumsq += x0[j] * x0[j] + x1[j] * x1[j];
            a[j]     = (_Float16)x0[j];
            a[8 + j] = (_Float16)x1[j];
        }

        // ---- prefetch next K-step's A values (overlaps the WMMAs) ----
        if (ks < NKS - 1) {
            const int cb = ((ks + 1) << 5) + (half << 4);
#pragma unroll
            for (int j = 0; j < 8; ++j) {
                x0[j] = qbase[(size_t)(cb + j)      * CSTR];
                x1[j] = qbase[(size_t)(cb + 8 + j)  * CSTR];
            }
        }

        // ---- 10 back-to-back WMMAs, no shared A/B registers ----
#pragma unroll
        for (int t = 0; t < NTILE; ++t)
            acc[t] = __builtin_amdgcn_wmma_f32_16x16x32_f16(
                false, a, false, bf[t], (short)0, acc[t], false, false);
    }

    // ---- combine row sumsq across lane halves (lanes l and l^16 share a row)
    sumsq += __shfl_xor(sumsq, 16, 32);
    const float invn = 1.0f / fmaxf(sqrtf(sumsq), 1e-12f);  // lane l: row l&15

    // k2 for this lane's 10 columns
    float myk2[NTILE];
#pragma unroll
    for (int t = 0; t < NTILE; ++t)
        myk2[t] = lds_k2[t * 16 + (lane & 15)];

    // ---- per-row argmin over 150 columns ----
    // C/D layout: VGPR r, lane l -> element (M = r + 8*(l>>4), N = l&15)
#pragma unroll
    for (int r = 0; r < 8; ++r) {
        const int rowLocal = r + (half << 3);
        const float rinv = __shfl(invn, rowLocal, 32);

        float best = 3.4e38f;
        int bestm = 0;
#pragma unroll
        for (int t = 0; t < NTILE; ++t) {
            const int m = t * 16 + (lane & 15);
            const float s = myk2[t] - 2.0f * acc[t][r] * rinv;
            if (m < NKEY && s < best) { best = s; bestm = m; }
        }
        // reduce across the 16 lanes of this half (xor stays within half)
#pragma unroll
        for (int off = 1; off < 16; off <<= 1) {
            float ov = __shfl_xor(best, off, 32);
            int   oi = __shfl_xor(bestm, off, 32);
            if (ov < best || (ov == best && oi < bestm)) { best = ov; bestm = oi; }
        }
        if ((lane & 15) == 0)
            idx_out[n0 + rowLocal] = bestm;
    }
    if (lane < 16)
        rinv_out[n0 + lane] = invn;
}

// ---------------------------------------------------------------------------
// Kernel 2: gathering_loss[n] = sum_c (q[n,c]*invnorm - keys[idx[n],c])^4
// One thread per pixel; query re-read hits L2 (128 MB < 192 MB L2),
// key rows (300 KB total) are cache-resident gathers.
// ---------------------------------------------------------------------------
__global__ void loss_kernel(const float* __restrict__ q,
                            const float* __restrict__ keys,
                            const int* __restrict__ idx,
                            const float* __restrict__ rinv,
                            float* __restrict__ out) {
    const int n = blockIdx.x * blockDim.x + threadIdx.x;
    if (n >= NPIX) return;
    const int b = n >> 12;
    const int p = n & 4095;
    const float* qb = q + (size_t)b * QB + p;
    const float* kr = keys + (size_t)idx[n] * CCH;
    const float s = rinv[n];
    float acc = 0.0f;
#pragma unroll 8
    for (int c = 0; c < CCH; ++c) {
        float d  = qb[(size_t)c * CSTR] * s - kr[c];
        float d2 = d * d;
        acc += d2 * d2;
    }
    out[n] = acc;
}

extern "C" void kernel_launch(void* const* d_in, const int* in_sizes, int n_in,
                              void* d_out, int out_size, void* d_ws, size_t ws_size,
                              hipStream_t stream) {
    const float* q    = (const float*)d_in[0];   // [16,512,64,64] f32
    const float* keys = (const float*)d_in[1];   // [150,512] f32
    float* out = (float*)d_out;                  // [16,64,64,1] f32

    int*   ws_idx  = (int*)d_ws;                           // 65536 ints
    float* ws_rinv = (float*)((char*)d_ws + NPIX * 4);     // 65536 floats

    const size_t smem = (size_t)NKS * MPAD * FRAG * 2 + (size_t)MPAD * 4; // 164480 B
    gemm_argmin_kernel<<<512, 256, smem, stream>>>(q, keys, ws_idx, ws_rinv);
    loss_kernel<<<NPIX / 256, 256, 0, stream>>>(q, keys, ws_idx, ws_rinv, out);
}
